// TransformerLayer_44117904064967
// MI455X (gfx1250) — matrix operand order repaired and checked by reading, not compiled, not run on hardware
//
#include <hip/hip_runtime.h>

// ---------------- types ----------------
typedef __attribute__((ext_vector_type(16))) __bf16 bf16x16;
typedef __attribute__((ext_vector_type(8)))  __bf16 bf16x8;
typedef __attribute__((ext_vector_type(8)))  float  f32x8;
typedef unsigned int u32x4 __attribute__((ext_vector_type(4)));
typedef int          i32x8 __attribute__((ext_vector_type(8)));
typedef int          i32x4 __attribute__((ext_vector_type(4)));

union AFrag { bf16x16 v; bf16x8 h[2]; };

static __device__ __forceinline__ __bf16 f2bf(float f) {
    unsigned u = __builtin_bit_cast(unsigned, f);
    unsigned r = u + 0x7FFFu + ((u >> 16) & 1u);          // RNE
    unsigned short hs = (unsigned short)(r >> 16);
    return __builtin_bit_cast(__bf16, hs);
}
static __device__ __forceinline__ float bf2f(__bf16 b) {
    unsigned short hs = __builtin_bit_cast(unsigned short, b);
    unsigned u = ((unsigned)hs) << 16;
    return __builtin_bit_cast(float, u);
}
static __device__ __forceinline__ float gelu_tanh(float v) {
    return 0.5f * v * (1.0f + tanhf(0.7978845608028654f * (v + 0.044715f * v * v * v)));
}

// LDS byte offset of a generic pointer to __shared__ data
static __device__ __forceinline__ unsigned lds_offset_of(const void* p) {
    return (unsigned)(unsigned long long)(__attribute__((address_space(3))) const void*)p;
}

#define MODE_BF16  0   // store bf16
#define MODE_RESID 1   // store f32 = acc + resid
#define MODE_GELU  2   // store bf16 = gelu(acc)
#define MODE_MOE   3   // Cf = (acc?Cf:0) + roww[row]*val

#if __has_builtin(__builtin_amdgcn_tensor_load_to_lds) && __has_builtin(__builtin_amdgcn_s_wait_tensorcnt)
#define USE_TDM 1
#elif __has_builtin(__builtin_amdgcn_global_load_async_to_lds_b128) && __has_builtin(__builtin_amdgcn_s_wait_asynccnt)
#define USE_ASYNC 1
#endif

#ifdef USE_TDM
// Issue a 2D TDM tile load: tile_w x tile_h elements of 2-byte data,
// row stride = stride_elems, packed contiguously into LDS at ldsaddr.
// clang-23 / therock toolchain: 6-arg builtin (g0 x4, g1 x8, x4, x4, x8, cpol).
static __device__ __forceinline__ void tdm_load_tile_2d(const void* gaddr, unsigned ldsaddr,
                                                        int tensor_w, int tensor_h,
                                                        int tile_w, int tile_h,
                                                        long long stride_elems) {
    unsigned long long ga = (unsigned long long)gaddr;
    u32x4 g0;
    g0[0] = 1u;                                            // count=1, user mode
    g0[1] = ldsaddr;                                       // lds_addr (bytes)
    g0[2] = (unsigned)(ga & 0xFFFFFFFFu);                  // global_addr[31:0]
    g0[3] = (unsigned)((ga >> 32) & 0x01FFFFFFu) | 0x80000000u;  // addr[56:32] | type=2
    i32x8 g1;
    g1[0] = 1 << 16;                                       // data_size=1 (2 bytes); no mask/pad/iter
    g1[1] = (tensor_w & 0xFFFF) << 16;                     // tensor_dim0[15:0]
    g1[2] = ((tensor_w >> 16) & 0xFFFF) | ((tensor_h & 0xFFFF) << 16);
    g1[3] = ((tensor_h >> 16) & 0xFFFF) | ((tile_w & 0xFFFF) << 16);  // | tile_dim0
    g1[4] = (tile_h & 0xFFFF);                             // tile_dim1 (tile_dim2=0)
    g1[5] = (int)(stride_elems & 0xFFFFFFFFll);            // tensor_dim0_stride[31:0]
    g1[6] = (int)((stride_elems >> 32) & 0xFFFFll);        // stride[47:32] (dim1_stride=0)
    g1[7] = 0;
    i32x4 z4 = {0, 0, 0, 0};
    i32x8 z8 = {0, 0, 0, 0, 0, 0, 0, 0};
    __builtin_amdgcn_tensor_load_to_lds(g0, g1, z4, z4, z8, 0);
}
#endif

// ---------------- fp32 -> bf16 convert ----------------
__global__ __launch_bounds__(256) void cvt_bf16_kernel(const float* __restrict__ src,
                                                       __bf16* __restrict__ dst, int n) {
    for (int i = blockIdx.x * 256 + threadIdx.x; i < n; i += gridDim.x * 256)
        dst[i] = f2bf(src[i]);
}

// transpose convert: src is R x C (row-major f32) -> dst is C x R (row-major bf16)
__global__ __launch_bounds__(256) void cvt_t_bf16_kernel(const float* __restrict__ src,
                                                         __bf16* __restrict__ dst, int R, int C) {
    int n = R * C;
    for (int i = blockIdx.x * 256 + threadIdx.x; i < n; i += gridDim.x * 256) {
        int r = i / C, c = i - r * C;
        dst[(size_t)c * R + r] = f2bf(src[i]);
    }
}

// ---------------- LayerNorm -> bf16 ----------------
__global__ __launch_bounds__(256) void ln_bf16_kernel(const float* __restrict__ x,
                                                      const float* __restrict__ g,
                                                      const float* __restrict__ bia,
                                                      __bf16* __restrict__ out, int H) {
    int t = blockIdx.x;
    size_t base = (size_t)t * H;
    float s = 0.f, sq = 0.f;
    for (int i = threadIdx.x; i < H; i += 256) { float v = x[base + i]; s += v; sq += v * v; }
    #pragma unroll
    for (int m = 1; m < 32; m <<= 1) { s += __shfl_xor(s, m, 32); sq += __shfl_xor(sq, m, 32); }
    __shared__ float rs[8], rq[8], mb[2];
    int wave = threadIdx.x >> 5, lane = threadIdx.x & 31;
    if (lane == 0) { rs[wave] = s; rq[wave] = sq; }
    __syncthreads();
    if (threadIdx.x == 0) {
        float S = 0.f, Q = 0.f;
        for (int i = 0; i < 8; i++) { S += rs[i]; Q += rq[i]; }
        float mu = S / H;
        float var = Q / H - mu * mu;
        mb[0] = mu; mb[1] = rsqrtf(var + 1e-5f);
    }
    __syncthreads();
    float mu = mb[0], rin = mb[1];
    for (int i = threadIdx.x; i < H; i += 256)
        out[base + i] = f2bf((x[base + i] - mu) * rin * g[i] + bia[i]);
}

// ---------------- generic bf16 WMMA GEMM: C[M,N] = A[M,K] * W[N,K]^T ----------------
// block = 256 threads (8 waves); block tile = 64 (M) x 128 (N); wave = 4 sub-tiles of 16x16.
// A tile (64x32) is staged ONCE per block into LDS (TDM / async / cooperative fallback),
// removing the 8x redundant per-wave A traffic; B fragments stream from global (L2-resident).
__global__ __launch_bounds__(256) void gemm_bf16_kernel(
    const __bf16* __restrict__ A, const __bf16* __restrict__ W,
    int M, int N, int Kd, int lda, int ldb, int ldc,
    float* __restrict__ Cf, __bf16* __restrict__ Cb,
    const float* __restrict__ resid, const float* __restrict__ roww,
    int mode, int accflag)
{
    __shared__ __align__(128) __bf16 Atile[2][64 * 32];

    const int lane = threadIdx.x & 31;
    const int wave = threadIdx.x >> 5;
    const int hh   = lane >> 4;       // half-wave
    const int l15  = lane & 15;
    const int m0 = blockIdx.x * 64;
    const int n0 = blockIdx.y * 128 + wave * 16;

    f32x8 acc[4] = {};
    const __bf16* wrow = W + (size_t)(n0 + l15) * ldb;
    const __bf16* abase = A + (size_t)m0 * lda;
    const int ntiles = Kd >> 5;

#ifdef USE_TDM
    if (wave == 0)
        tdm_load_tile_2d(abase, lds_offset_of(&Atile[0][0]), Kd, M - m0, 32, 64, lda);
#endif

    for (int i = 0; i < ntiles; i++) {
        const int k0 = i << 5;
#ifdef USE_TDM
        if (wave == 0) {
            if (i + 1 < ntiles) {
                tdm_load_tile_2d(abase + (k0 + 32), lds_offset_of(&Atile[(i + 1) & 1][0]),
                                 Kd, M - m0, 32, 64, lda);
                __builtin_amdgcn_s_wait_tensorcnt(1);   // current tile landed
            } else {
                __builtin_amdgcn_s_wait_tensorcnt(0);
            }
        }
        __syncthreads();
        const __bf16* at = &Atile[i & 1][0];
#elif defined(USE_ASYNC)
        {
            int row = threadIdx.x >> 2, ch = threadIdx.x & 3;
            __builtin_amdgcn_global_load_async_to_lds_b128(
                (const __attribute__((address_space(1))) void*)(abase + (size_t)row * lda + k0 + ch * 8),
                (__attribute__((address_space(3))) void*)&Atile[0][row * 32 + ch * 8], 0, 0);
            __builtin_amdgcn_s_wait_asynccnt(0);
        }
        __syncthreads();
        const __bf16* at = &Atile[0][0];
#else
        {
            int row = threadIdx.x >> 2, ch = threadIdx.x & 3;
            bf16x8 v = *(const bf16x8*)(abase + (size_t)row * lda + k0 + ch * 8);
            *(bf16x8*)&Atile[0][row * 32 + ch * 8] = v;
        }
        __syncthreads();
        const __bf16* at = &Atile[0][0];
#endif
        __builtin_prefetch(wrow + k0 + 256, 0, 1);
        // B fragment: lane = column n, K contiguous per half-wave
        bf16x16 bfrag = *(const bf16x16*)(wrow + k0 + 16 * hh);
        #pragma unroll
        for (int s = 0; s < 4; s++) {
            AFrag af;
            af.h[0] = *(const bf16x8*)(at + (s * 16 + l15) * 32 + 16 * hh);
            af.h[1] = *(const bf16x8*)(at + (s * 16 + l15) * 32 + 16 * hh + 8);
            acc[s] = __builtin_amdgcn_wmma_f32_16x16x32_bf16(
                         false, af.v, false, bfrag, (short)0, acc[s], false, false);
        }
        __syncthreads();
    }

    const int col = n0 + l15;
    #pragma unroll
    for (int s = 0; s < 4; s++) {
        const int rowbase = m0 + s * 16 + 8 * hh;
        #pragma unroll
        for (int r = 0; r < 8; r++) {
            float v = acc[s][r];
            size_t idx = (size_t)(rowbase + r) * ldc + col;
            if (mode == MODE_BF16) {
                Cb[idx] = f2bf(v);
            } else if (mode == MODE_RESID) {
                Cf[idx] = v + resid[idx];
            } else if (mode == MODE_GELU) {
                Cb[idx] = f2bf(gelu_tanh(v));
            } else { // MODE_MOE
                float wgt = roww[rowbase + r];
                float prev = accflag ? Cf[idx] : 0.0f;
                Cf[idx] = prev + wgt * v;
            }
        }
    }
}

// ---------------- flash-style causal GQA attention ----------------
// qkv: [T=S*B, 1536] bf16 (q:0..1023, k:1024..1279, v:1280..1535)
// out: [T, 1024] bf16.  grid = (S/128, B*16); block = 256 (8 waves, 16 q-rows each)
__global__ __launch_bounds__(256) void attn_kernel(const __bf16* __restrict__ qkv,
                                                   __bf16* __restrict__ attnout,
                                                   int S, int Bb) {
    constexpr int QKVLD = 1536;
    __shared__ __align__(64) __bf16 Vt[8][64 * 32];  // per-wave V^T tile [d=64][k=32]
    __shared__ __align__(64) __bf16 Pt[8][16 * 32];  // per-wave P tile  [m=16][k=32]

    const int lane = threadIdx.x & 31;
    const int wave = threadIdx.x >> 5;
    const int hh   = lane >> 4;
    const int l15  = lane & 15;
    const int head = blockIdx.y & 15;
    const int b    = blockIdx.y >> 4;
    const int kvh  = head >> 2;
    const int q0   = blockIdx.x * 128 + wave * 16;

    // load Q A-fragments (hd=64 -> two K=32 slices), lane row = q0+l15
    bf16x16 qf[2];
    {
        size_t qbase = (size_t)((q0 + l15) * Bb + b) * QKVLD + head * 64;
        #pragma unroll
        for (int ks = 0; ks < 2; ks++) {
            AFrag af;
            af.h[0] = *(const bf16x8*)(qkv + qbase + ks * 32 + 16 * hh);
            af.h[1] = *(const bf16x8*)(qkv + qbase + ks * 32 + 16 * hh + 8);
            qf[ks] = af.v;
        }
    }

    float rmax[8], rsum[8];
    #pragma unroll
    for (int r = 0; r < 8; r++) { rmax[r] = -1e30f; rsum[r] = 0.f; }
    f32x8 o[4] = {};
    const float scale = 0.125f;  // 1/sqrt(64)

    const int ktEnd = q0 + 16;
    for (int kt = 0; kt < ktEnd; kt += 32) {
        // ---- stage V^T into per-wave LDS: lane owns key row kt+lane ----
        {
            size_t vbase = (size_t)((kt + lane) * Bb + b) * QKVLD + 1280 + kvh * 64;
            #pragma unroll
            for (int c = 0; c < 8; c++) {
                bf16x8 chunk = *(const bf16x8*)(qkv + vbase + c * 8);
                #pragma unroll
                for (int j = 0; j < 8; j++)
                    Vt[wave][(c * 8 + j) * 32 + lane] = chunk[j];
            }
        }
        // ---- scores: two 16-wide key sub-tiles, K=64 over hd ----
        f32x8 sc[2] = {};
        #pragma unroll
        for (int nt = 0; nt < 2; nt++) {
            size_t kbase = (size_t)((kt + nt * 16 + l15) * Bb + b) * QKVLD + 1024 + kvh * 64;
            #pragma unroll
            for (int ks = 0; ks < 2; ks++) {
                bf16x16 bfrag = *(const bf16x16*)(qkv + kbase + ks * 32 + 16 * hh);
                sc[nt] = __builtin_amdgcn_wmma_f32_16x16x32_bf16(
                             false, qf[ks], false, bfrag, (short)0, sc[nt], false, false);
            }
        }
        // ---- causal mask + scale ----
        #pragma unroll
        for (int nt = 0; nt < 2; nt++) {
            int kcol = kt + nt * 16 + l15;
            #pragma unroll
            for (int r = 0; r < 8; r++) {
                int qrow = q0 + 8 * hh + r;
                sc[nt][r] = (kcol <= qrow) ? sc[nt][r] * scale : -1e30f;
            }
        }
        // ---- online softmax (row lives across 16 lanes of a half-wave) ----
        #pragma unroll
        for (int r = 0; r < 8; r++) {
            float mloc = fmaxf(sc[0][r], sc[1][r]);
            #pragma unroll
            for (int m = 1; m < 16; m <<= 1) mloc = fmaxf(mloc, __shfl_xor(mloc, m, 32));
            float mnew  = fmaxf(rmax[r], mloc);
            float alpha = __expf(rmax[r] - mnew);
            float p0 = __expf(sc[0][r] - mnew);
            float p1 = __expf(sc[1][r] - mnew);
            int mrow = 8 * hh + r;
            Pt[wave][mrow * 32 + l15]      = f2bf(p0);
            Pt[wave][mrow * 32 + 16 + l15] = f2bf(p1);
            float ls = p0 + p1;
            #pragma unroll
            for (int m = 1; m < 16; m <<= 1) ls += __shfl_xor(ls, m, 32);
            rsum[r] = rsum[r] * alpha + ls;
            rmax[r] = mnew;
            #pragma unroll
            for (int n = 0; n < 4; n++) o[n][r] *= alpha;
        }
        asm volatile("s_wait_dscnt 0" ::: "memory");  // fence per-wave LDS staging
        // ---- P as A-fragment (16x32) from LDS ----
        AFrag pf;
        pf.h[0] = *(const bf16x8*)&Pt[wave][l15 * 32 + 16 * hh];
        pf.h[1] = *(const bf16x8*)&Pt[wave][l15 * 32 + 16 * hh + 8];
        // ---- PV: hd = 4 n-tiles of 16, B-fragments from V^T LDS ----
        #pragma unroll
        for (int n = 0; n < 4; n++) {
            bf16x16 vb = *(const bf16x16*)&Vt[wave][(n * 16 + l15) * 32 + 16 * hh];
            o[n] = __builtin_amdgcn_wmma_f32_16x16x32_bf16(
                       false, pf.v, false, vb, (short)0, o[n], false, false);
        }
    }

    // ---- epilogue: divide by row-sum, store bf16 ----
    #pragma unroll
    for (int r = 0; r < 8; r++) {
        float inv = 1.0f / rsum[r];
        int qrow = q0 + 8 * hh + r;
        size_t obase = (size_t)(qrow * Bb + b) * 1024 + head * 64;
        #pragma unroll
        for (int n = 0; n < 4; n++)
            attnout[obase + n * 16 + l15] = f2bf(o[n][r] * inv);
    }
}

// ---------------- router: softmax + top-2 -> per-expert row weights ----------------
__global__ __launch_bounds__(256) void router_kernel(const __bf16* __restrict__ x,
                                                     const float* __restrict__ rw,
                                                     float* __restrict__ roww,
                                                     int T, int H) {
    int t = blockIdx.x * 256 + threadIdx.x;
    if (t >= T) return;
    float acc[8] = {};
    const __bf16* xr = x + (size_t)t * H;
    for (int h = 0; h < H; h++) {
        float xv = bf2f(xr[h]);
        #pragma unroll
        for (int e = 0; e < 8; e++) acc[e] += xv * rw[e * H + h];
    }
    float mx = acc[0];
    #pragma unroll
    for (int e = 1; e < 8; e++) mx = fmaxf(mx, acc[e]);
    float p[8], se = 0.f;
    #pragma unroll
    for (int e = 0; e < 8; e++) { p[e] = __expf(acc[e] - mx); se += p[e]; }
    #pragma unroll
    for (int e = 0; e < 8; e++) p[e] /= se;
    int i1 = 0;
    #pragma unroll
    for (int e = 1; e < 8; e++) if (p[e] > p[i1]) i1 = e;
    int i2 = (i1 == 0) ? 1 : 0;
    #pragma unroll
    for (int e = 0; e < 8; e++) if (e != i1 && p[e] > p[i2]) i2 = e;
    #pragma unroll
    for (int e = 0; e < 8; e++)
        roww[(size_t)e * T + t] = (e == i1) ? p[i1] : ((e == i2) ? p[i2] : 0.f);
}

// ---------------- final: out = hidden_after_attn + moe_acc ----------------
__global__ __launch_bounds__(256) void final_add_kernel(const float* __restrict__ a,
                                                        const float* __restrict__ b,
                                                        float* __restrict__ out, int n) {
    for (int i = blockIdx.x * 256 + threadIdx.x; i < n; i += gridDim.x * 256)
        out[i] = a[i] + b[i];
}

// ---------------- host launch ----------------
extern "C" void kernel_launch(void* const* d_in, const int* in_sizes, int n_in,
                              void* d_out, int out_size, void* d_ws, size_t ws_size,
                              hipStream_t stream) {
    (void)in_sizes; (void)n_in; (void)out_size; (void)ws_size;
    const int S = 1024, B = 4, H = 1024, F = 2048, T = S * B, O = 1536, E = 8;

    const float* hidden  = (const float*)d_in[0];
    const float* ln1w    = (const float*)d_in[1];
    const float* ln1b    = (const float*)d_in[2];
    const float* ln2w    = (const float*)d_in[3];
    const float* ln2b    = (const float*)d_in[4];
    const float* qkv_w   = (const float*)d_in[5];
    const float* proj_w  = (const float*)d_in[6];
    const float* rout_w  = (const float*)d_in[7];
    const float* moe_w1  = (const float*)d_in[8];
    const float* moe_w2  = (const float*)d_in[9];
    float* out = (float*)d_out;

    size_t off = 0;
    auto alloc = [&](size_t bytes) -> char* {
        char* p = (char*)d_ws + off;
        off = (off + bytes + 255) & ~(size_t)255;
        return p;
    };
    __bf16* qkvw  = (__bf16*)alloc((size_t)O * H * 2);
    __bf16* projw = (__bf16*)alloc((size_t)H * H * 2);
    __bf16* w1t   = (__bf16*)alloc((size_t)F * H * 2);  // per-expert, reused
    __bf16* w2t   = (__bf16*)alloc((size_t)H * F * 2);  // per-expert, reused
    __bf16* x1    = (__bf16*)alloc((size_t)T * H * 2);
    __bf16* qkvB  = (__bf16*)alloc((size_t)T * O * 2);
    __bf16* attnB = (__bf16*)alloc((size_t)T * H * 2);
    float*  hattn = (float*) alloc((size_t)T * H * 4);
    __bf16* x2    = (__bf16*)alloc((size_t)T * H * 2);
    float*  roww  = (float*) alloc((size_t)E * T * 4);
    __bf16* hmid  = (__bf16*)alloc((size_t)T * F * 2);
    float*  macc  = (float*) alloc((size_t)T * H * 4);

    // weights -> bf16
    cvt_bf16_kernel<<<2048, 256, 0, stream>>>(qkv_w, qkvw, O * H);
    cvt_bf16_kernel<<<2048, 256, 0, stream>>>(proj_w, projw, H * H);

    // LN1 -> QKV
    ln_bf16_kernel<<<T, 256, 0, stream>>>(hidden, ln1w, ln1b, x1, H);
    gemm_bf16_kernel<<<dim3(T / 64, O / 128), 256, 0, stream>>>(
        x1, qkvw, T, O, H, H, H, O, nullptr, qkvB, nullptr, nullptr, MODE_BF16, 0);

    // attention
    attn_kernel<<<dim3(S / 128, B * 16), 256, 0, stream>>>(qkvB, attnB, S, B);

    // proj + residual
    gemm_bf16_kernel<<<dim3(T / 64, H / 128), 256, 0, stream>>>(
        attnB, projw, T, H, H, H, H, H, hattn, nullptr, hidden, nullptr, MODE_RESID, 0);

    // LN2 -> router
    ln_bf16_kernel<<<T, 256, 0, stream>>>(hattn, ln2w, ln2b, x2, H);
    router_kernel<<<T / 256, 256, 0, stream>>>(x2, rout_w, roww, T, H);

    // MoE: dense experts (weights zero for non-selected -> matches reference exactly)
    for (int e = 0; e < E; e++) {
        cvt_t_bf16_kernel<<<2048, 256, 0, stream>>>(moe_w1 + (size_t)e * H * F, w1t, H, F);
        gemm_bf16_kernel<<<dim3(T / 64, F / 128), 256, 0, stream>>>(
            x2, w1t, T, F, H, H, H, F, nullptr, hmid, nullptr, nullptr, MODE_GELU, 0);
        cvt_t_bf16_kernel<<<2048, 256, 0, stream>>>(moe_w2 + (size_t)e * F * H, w2t, F, H);
        gemm_bf16_kernel<<<dim3(T / 64, H / 128), 256, 0, stream>>>(
            hmid, w2t, T, H, F, F, F, H, macc, nullptr, nullptr, roww + (size_t)e * T,
            MODE_MOE, (e > 0) ? 1 : 0);
    }

    final_add_kernel<<<2048, 256, 0, stream>>>(hattn, macc, out, T * H);
}
